// ContextTransformer_10445360463875
// MI455X (gfx1250) — matrix operand order repaired and checked by reading, not compiled, not run on hardware
//
#include <hip/hip_runtime.h>

#define BB   8
#define TTn  512
#define DOFn 217
#define NLn  6
#define NHn  8
#define DMn  512
#define DFFn 2048
#define HDn  64
#define ROWS (BB*TTn)   // 4096

typedef __attribute__((ext_vector_type(16))) __bf16 v16bf;
typedef __attribute__((ext_vector_type(8)))  float  v8f;

union FragB16 { v16bf v; unsigned int d[8]; unsigned short u[16]; };
union FragF32 { v8f   v; float f[8]; };
union U128    { uint4 q; unsigned short s[8]; };

__device__ inline unsigned short f2bf(float x) {
  unsigned int u = __float_as_uint(x);
  u += 0x7FFFu + ((u >> 16) & 1u);              // round-to-nearest-even
  return (unsigned short)(u >> 16);
}
__device__ inline unsigned int pack2bf(float a, float b) {
  return (unsigned int)f2bf(a) | ((unsigned int)f2bf(b) << 16);
}
// A-fragment (16x32 bf16) per-lane dword K pattern from ISA 7.12.2
__device__ inline int kpat(int i, int hlf) {
  return ((i < 4) ? (2*i) : (16 + 2*(i-4))) + 8*hlf;
}
__device__ inline v8f wmma_bf(v16bf a, v16bf b, v8f c) {
  return __builtin_amdgcn_wmma_f32_16x16x32_bf16(false, a, false, b, (short)0, c, false, false);
}

// ---------------- weight convert f32 -> bf16 with K padding -------------
__global__ __launch_bounds__(256) void cvt_w_kernel(const float* __restrict__ src,
                                                    unsigned short* __restrict__ dst,
                                                    int N, int K, int Kp) {
  int t = blockIdx.x*256 + threadIdx.x;
  if (t >= N*Kp) return;
  int n = t / Kp, kk = t - n*Kp;
  dst[t] = (kk < K) ? f2bf(src[(size_t)n*K + kk]) : (unsigned short)0;
}

// ---------------- generic WMMA GEMM: Y = act(X*W^T + b) [+ add] --------
// Requirements: M % 128 == 0, Kp % 64 == 0, ldx % 4 == 0, X 16B-aligned.
__global__ __launch_bounds__(256) void gemm_kernel(
    const float* __restrict__ X, int ldx,
    const unsigned short* __restrict__ Wb,   // [N][Kp] bf16
    const float* __restrict__ bias,
    float* __restrict__ Y, int ldy,
    const float* __restrict__ addsrc, int ldadd,
    const float* __restrict__ alpha, int act,
    int M, int N, int Kp) {
  __shared__ unsigned short sA[128][72];     // [m][k] bf16, K-depth 64
  __shared__ unsigned short sB[64][72];      // [k][n] bf16  (W transposed)
  int tid  = threadIdx.x;
  int lane = tid & 31, wave = tid >> 5;
  int wm = wave & 3, wn = wave >> 2;         // 4x2 wave grid -> 128x64 tile
  int mrow = lane & 15, hlf = lane >> 4;
  int m0 = blockIdx.y * 128;
  int n0 = blockIdx.x * 64;
  v8f zero8 = {0.f,0.f,0.f,0.f,0.f,0.f,0.f,0.f};
  FragF32 acc[2][2];
  for (int si=0; si<2; si++) for (int sj=0; sj<2; sj++) acc[si][sj].v = zero8;

  for (int k0 = 0; k0 < Kp; k0 += 64) {
    __syncthreads();
    // stage A: 128 x 64 f32 -> bf16, vectorized float4 (2048 loads / block)
    for (int t = tid; t < 2048; t += 256) {
      int r  = t >> 4;                 // 16 float4 per row
      int kk = (t & 15) * 4;
      const float* xp = &X[(size_t)(m0 + r)*ldx + k0 + kk];
      float4 xv = *(const float4*)xp;
      *(unsigned int*)&sA[r][kk]     = pack2bf(xv.x, xv.y);
      *(unsigned int*)&sA[r][kk + 2] = pack2bf(xv.z, xv.w);
      if (k0 + 64 < Kp) __builtin_prefetch(xp + 64, 0, 0);   // global_prefetch
    }
    // stage B transposed: 64 n x 64 k bf16, 16-byte loads (512 / block)
    for (int t = tid; t < 512; t += 256) {
      int nn = t >> 3;
      int kk = (t & 7) * 8;
      int gn = n0 + nn;
      U128 wv;
      if (gn < N) {
        wv.q = *(const uint4*)&Wb[(size_t)gn*Kp + k0 + kk];
        if (k0 + 64 < Kp) __builtin_prefetch(&Wb[(size_t)gn*Kp + k0 + 64 + kk], 0, 0);
      } else {
        wv.q = make_uint4(0u, 0u, 0u, 0u);
      }
      #pragma unroll
      for (int i = 0; i < 8; i++) sB[kk + i][nn] = wv.s[i];
    }
    __syncthreads();
    FragB16 af[2][2], bfr[2][2];
    #pragma unroll
    for (int s = 0; s < 2; s++) {
      int rb = wm*32 + s*16 + mrow;
      #pragma unroll
      for (int c = 0; c < 2; c++)
        #pragma unroll
        for (int i = 0; i < 8; i++)
          af[s][c].d[i] = *(const unsigned int*)&sA[rb][c*32 + kpat(i,hlf)];
    }
    #pragma unroll
    for (int s = 0; s < 2; s++) {
      int nb = wn*32 + s*16;
      #pragma unroll
      for (int c = 0; c < 2; c++)
        #pragma unroll
        for (int i = 0; i < 8; i++)
          { bfr[s][c].u[2*i] = sB[c*32 + kpat(i,hlf)][nb + mrow]; bfr[s][c].u[2*i+1] = sB[c*32 + kpat(i,hlf) + 1][nb + mrow]; }
    }
    #pragma unroll
    for (int si = 0; si < 2; si++)
      #pragma unroll
      for (int sj = 0; sj < 2; sj++)
        #pragma unroll
        for (int c = 0; c < 2; c++)
          acc[si][sj].v = wmma_bf(af[si][c].v, bfr[sj][c].v, acc[si][sj].v);
  }
  float al = act ? alpha[0] : 0.f;
  #pragma unroll
  for (int si = 0; si < 2; si++)
    #pragma unroll
    for (int sj = 0; sj < 2; sj++)
      #pragma unroll
      for (int i = 0; i < 8; i++) {
        int row = m0 + wm*32 + si*16 + i + (hlf ? 8 : 0);   // C layout: M in VGPR idx
        int col = n0 + wn*32 + sj*16 + mrow;
        if (col < N) {
          float tv = acc[si][sj].f[i] + bias[col];
          if (act)    tv = (tv >= 0.f) ? tv : al*tv;
          if (addsrc) tv += addsrc[(size_t)row*ldadd + col];
          Y[(size_t)row*ldy + col] = tv;
        }
      }
}

// ---------------- layernorm (shared gamma/beta), one wave per row ------
__global__ __launch_bounds__(256) void ln_kernel(const float* __restrict__ xin,
                                                 const float* __restrict__ g,
                                                 const float* __restrict__ bv,
                                                 float* __restrict__ yout, int rows) {
  int wave = blockIdx.x*8 + (threadIdx.x >> 5);
  int lane = threadIdx.x & 31;
  if (wave >= rows) return;
  const float* xr = xin + (size_t)wave*DMn;
  float vals[16], s = 0.f, s2 = 0.f;
  #pragma unroll
  for (int i = 0; i < 4; i++) {
    float4 xv = *(const float4*)&xr[lane*4 + 128*i];
    vals[4*i+0]=xv.x; vals[4*i+1]=xv.y; vals[4*i+2]=xv.z; vals[4*i+3]=xv.w;
    s += xv.x+xv.y+xv.z+xv.w;
    s2 += xv.x*xv.x+xv.y*xv.y+xv.z*xv.z+xv.w*xv.w;
  }
  #pragma unroll
  for (int o = 16; o >= 1; o >>= 1) { s += __shfl_xor(s, o, 32); s2 += __shfl_xor(s2, o, 32); }
  float m = s * (1.f/DMn);
  float var = s2 * (1.f/DMn) - m*m;
  float r = rsqrtf(var + 1e-8f);
  float* yr = yout + (size_t)wave*DMn;
  #pragma unroll
  for (int i = 0; i < 4; i++) {
    int c = lane*4 + 128*i;
    float4 o4;
    o4.x = (vals[4*i+0]-m)*r*g[c+0] + bv[c+0];
    o4.y = (vals[4*i+1]-m)*r*g[c+1] + bv[c+1];
    o4.z = (vals[4*i+2]-m)*r*g[c+2] + bv[c+2];
    o4.w = (vals[4*i+3]-m)*r*g[c+3] + bv[c+3];
    *(float4*)&yr[c] = o4;
  }
}

// ---------------- flash attention with WMMA + relative positions -------
// Q,K,V,O: [B*T][512] f32 (head h at cols h*64..). ET: [64][1024] bf16 = E_rel^T
__global__ __launch_bounds__(256) void attn_kernel(
    const float* __restrict__ Q, const float* __restrict__ Kp,
    const float* __restrict__ V, const unsigned short* __restrict__ ET,
    const float* __restrict__ maskin, float* __restrict__ O) {
  __shared__ unsigned short sKT[64][34];     // [hd][j]  (K^T chunk)
  __shared__ unsigned short sV[32][68];      // [j][hd]
  __shared__ unsigned short sP[8][16][34];   // per-wave P tile bounce
  int b = blockIdx.x >> 3, hh = blockIdx.x & 7;
  int wave = threadIdx.x >> 5, lane = threadIdx.x & 31;
  int mrow = lane & 15, hlf = lane >> 4;
  int i0 = blockIdx.y*128 + wave*16;
  const float* Qb = Q + ((size_t)b*TTn)*DMn + hh*HDn;
  FragB16 qa[2];                             // Q A-frags, K-dim = 64
  #pragma unroll
  for (int c = 0; c < 2; c++)
    #pragma unroll
    for (int i = 0; i < 8; i++) {
      int K0 = c*32 + kpat(i, hlf);
      float2 qv = *(const float2*)&Qb[(size_t)(i0+mrow)*DMn + K0];
      qa[c].d[i] = pack2bf(qv.x, qv.y);
    }
  v8f zero8 = {0.f,0.f,0.f,0.f,0.f,0.f,0.f,0.f};
  FragF32 of[4];
  #pragma unroll
  for (int t = 0; t < 4; t++) of[t].v = zero8;
  float mprev[8], lsum[8];
  #pragma unroll
  for (int i = 0; i < 8; i++) { mprev[i] = -1e30f; lsum[i] = 0.f; }

  for (int j0 = 0; j0 < TTn; j0 += 32) {
    __syncthreads();
    // stage K^T and V chunks: 32x64 f32 each, float4-vectorized
    for (int t = threadIdx.x; t < 512; t += 256) {
      int jj = t >> 4, kd = (t & 15) * 4;
      size_t gr = ((size_t)b*TTn + j0 + jj)*DMn + hh*HDn + kd;
      float4 kv = *(const float4*)&Kp[gr];
      float4 vv = *(const float4*)&V[gr];
      sKT[kd  ][jj] = f2bf(kv.x);
      sKT[kd+1][jj] = f2bf(kv.y);
      sKT[kd+2][jj] = f2bf(kv.z);
      sKT[kd+3][jj] = f2bf(kv.w);
      *(unsigned int*)&sV[jj][kd]     = pack2bf(vv.x, vv.y);
      *(unsigned int*)&sV[jj][kd + 2] = pack2bf(vv.z, vv.w);
      if (j0 + 32 < TTn) {
        __builtin_prefetch(&Kp[gr + (size_t)32*DMn], 0, 0);
        __builtin_prefetch(&V[gr + (size_t)32*DMn], 0, 0);
      }
    }
    __syncthreads();
    // S = Q*K^T  (two 16x16 subtiles over 32 keys)
    FragF32 sf[2]; sf[0].v = zero8; sf[1].v = zero8;
    #pragma unroll
    for (int st = 0; st < 2; st++)
      #pragma unroll
      for (int c = 0; c < 2; c++) {
        FragB16 kb;
        #pragma unroll
        for (int i = 0; i < 8; i++) { kb.u[2*i] = sKT[c*32+kpat(i,hlf)][st*16 + mrow]; kb.u[2*i+1] = sKT[c*32+kpat(i,hlf)+1][st*16 + mrow]; }
        sf[st].v = wmma_bf(qa[c].v, kb.v, sf[st].v);
      }
    // QE windows: contiguous 16-row slices of E_rel starting at r0, r0+16, r0+32
    int r0 = j0 - i0 + 496;
    FragF32 qf[3];
    #pragma unroll
    for (int w = 0; w < 3; w++) {
      qf[w].v = zero8;
      int rb = r0 + 16*w;
      #pragma unroll
      for (int c = 0; c < 2; c++) {
        FragB16 eb;
        #pragma unroll
        for (int i = 0; i < 8; i++)
          { eb.u[2*i] = ET[(size_t)(c*32+kpat(i,hlf))*1024 + rb + mrow]; eb.u[2*i+1] = ET[(size_t)(c*32+kpat(i,hlf)+1)*1024 + rb + mrow]; }
        qf[w].v = wmma_bf(qa[c].v, eb.v, qf[w].v);
      }
    }
    float mc[2];
    #pragma unroll
    for (int st = 0; st < 2; st++) {
      int j = j0 + st*16 + mrow;
      mc[st] = (maskin[(size_t)b*TTn + j] == 0.f) ? -1e8f : 0.f;
    }
    // diagonal gather of srel + scale + mask
    float sv[2][8];
    #pragma unroll
    for (int st = 0; st < 2; st++)
      #pragma unroll
      for (int i = 0; i < 8; i++) {
        int m = i + (hlf ? 8 : 0);
        int w = mrow - m + 15;                 // 0..30 window index
        int src = (w & 15) + (hlf ? 16 : 0);
        float g0 = __shfl(qf[st].f[i],   src, 32);
        float g1 = __shfl(qf[st+1].f[i], src, 32);
        float srel = (w < 16) ? g0 : g1;
        sv[st][i] = (sf[st].f[i] + srel) * 0.125f + mc[st];
      }
    // online softmax over 32-key chunk
    float p0a[8], p1a[8], scl[8];
    #pragma unroll
    for (int i = 0; i < 8; i++) {
      float rm = fmaxf(sv[0][i], sv[1][i]);
      #pragma unroll
      for (int o = 8; o >= 1; o >>= 1) rm = fmaxf(rm, __shfl_xor(rm, o, 32));
      float mn = fmaxf(mprev[i], rm);
      scl[i] = __expf(mprev[i] - mn);
      float p0 = __expf(sv[0][i] - mn);
      float p1 = __expf(sv[1][i] - mn);
      float rs = p0 + p1;
      #pragma unroll
      for (int o = 8; o >= 1; o >>= 1) rs += __shfl_xor(rs, o, 32);
      lsum[i] = lsum[i]*scl[i] + rs;
      mprev[i] = mn;
      p0a[i] = p0; p1a[i] = p1;
    }
    #pragma unroll
    for (int t = 0; t < 4; t++)
      #pragma unroll
      for (int i = 0; i < 8; i++) of[t].f[i] *= scl[i];
    // bounce P through wave-private LDS into A-fragment layout
    #pragma unroll
    for (int i = 0; i < 8; i++) {
      int m = i + (hlf ? 8 : 0);
      sP[wave][m][mrow]      = f2bf(p0a[i]);
      sP[wave][m][mrow + 16] = f2bf(p1a[i]);
    }
    FragB16 pa;
    #pragma unroll
    for (int i = 0; i < 8; i++) pa.d[i] = *(const unsigned int*)&sP[wave][mrow][kpat(i,hlf)];
    // O += P * V
    #pragma unroll
    for (int t = 0; t < 4; t++) {
      FragB16 vb;
      #pragma unroll
      for (int i = 0; i < 8; i++) { vb.u[2*i] = sV[kpat(i,hlf)][t*16 + mrow]; vb.u[2*i+1] = sV[kpat(i,hlf)+1][t*16 + mrow]; }
      of[t].v = wmma_bf(pa.v, vb.v, of[t].v);
    }
  }
  #pragma unroll
  for (int t = 0; t < 4; t++)
    #pragma unroll
    for (int i = 0; i < 8; i++) {
      int row = i0 + i + (hlf ? 8 : 0);
      int col = hh*HDn + t*16 + mrow;
      O[((size_t)b*TTn + row)*DMn + col] = of[t].f[i] / lsum[i];
    }
}

// ---------------- small helper kernels --------------------------------
__global__ __launch_bounds__(256) void concat_kernel(const float* __restrict__ x,
                                                     const float* __restrict__ mask,
                                                     float* __restrict__ xc) {
  int t = blockIdx.x*256 + threadIdx.x;
  if (t >= ROWS*256) return;
  int r = t >> 8, c = t & 255;
  xc[t] = (c < DOFn) ? x[(size_t)r*DOFn + c] : ((c == DOFn) ? mask[r] : 0.f);
}

__global__ __launch_bounds__(256) void kf1_kernel(const float* __restrict__ pkf,
                                                  const float* __restrict__ w,
                                                  const float* __restrict__ bv,
                                                  const float* __restrict__ a,
                                                  float* __restrict__ y) {
  int t = blockIdx.x*256 + threadIdx.x;
  if (t >= ROWS*DMn) return;
  int r = t >> 9, n = t & 511;
  float vv = w[n*2]*pkf[(size_t)r*2] + w[n*2+1]*pkf[(size_t)r*2+1] + bv[n];
  float al = a[0];
  y[t] = (vv >= 0.f) ? vv : al*vv;
}

// t = arange(-511..511); rows padded to 1024, row 1023 zeroed
__global__ __launch_bounds__(256) void rel1_kernel(const float* __restrict__ w,
                                                   const float* __restrict__ bv,
                                                   const float* __restrict__ a,
                                                   float* __restrict__ y) {
  int t = blockIdx.x*256 + threadIdx.x;
  if (t >= 1024*DMn) return;
  int r = t >> 9, n = t & 511;
  float out = 0.f;
  if (r < 1023) {
    float tv = (float)(r - (TTn - 1));
    float vv = w[n]*tv + bv[n];
    float al = a[0];
    out = (vv >= 0.f) ? vv : al*vv;
  }
  y[t] = out;
}

// E_rel [1024][64] f32 -> ET [64][1024] bf16 (row 1023 zero-padded)
__global__ __launch_bounds__(256) void build_et_kernel(const float* __restrict__ erel,
                                                       unsigned short* __restrict__ et) {
  int t = blockIdx.x*256 + threadIdx.x;
  if (t >= 64*1024) return;
  int kd = t >> 10, r = t & 1023;
  et[t] = (r < 1023) ? f2bf(erel[(size_t)r*64 + kd]) : (unsigned short)0;
}

__global__ __launch_bounds__(256) void final_kernel(const float* __restrict__ y,
                                                    float* __restrict__ out) {
  int t = blockIdx.x*256 + threadIdx.x;
  if (t >= ROWS*221) return;
  int r = t / 221, c = t - r*221;
  float vv = y[t];
  if (c < DOFn) out[(size_t)r*DOFn + c] = vv;
  else          out[(size_t)ROWS*DOFn + (size_t)r*4 + (c - DOFn)] = 1.f/(1.f + __expf(-vv));
}

// ---------------- host orchestration ----------------------------------
extern "C" void kernel_launch(void* const* d_in, const int* in_sizes, int n_in,
                              void* d_out, int out_size, void* d_ws, size_t ws_size,
                              hipStream_t stream) {
  (void)in_sizes; (void)n_in; (void)out_size; (void)ws_size;
  // jax pytree flatten (dicts sorted by key):
  // 0:x 1:mask_in 2:p_kf | 3:dec.a 4:dec.l1.b 5:dec.l1.w 6:dec.l2.b 7:dec.l2.w
  // 8:enc.a 9:enc.l1.b 10:enc.l1.w 11:enc.l2.b 12:enc.l2.w
  // 13:kf.a 14:kf.l1.b 15:kf.l1.w 16:kf.l2.b 17:kf.l2.w
  // 18.. layers (13 each): k.b k.w o.b o.w q.b q.w v.b v.w ffn.a l1.b l1.w l2.b l2.w
  // 96:ln.b 97:ln.g 98:rel.a 99:rel.l1.b 100:rel.l1.w 101:rel.l2.b 102:rel.l2.w
  const int L0 = 18, LS = 13;
  const int LN_B = L0 + NLn*LS, LN_G = LN_B+1, REL_A = LN_B+2,
            REL_L1B = LN_B+3, REL_L1W = LN_B+4, REL_L2B = LN_B+5, REL_L2W = LN_B+6;
  auto F = [&](int i) { return (const float*)d_in[i]; };
  const float* x      = F(0);
  const float* maskin = F(1);
  const float* pkf    = F(2);

  char* wp = (char*)d_ws;
  auto bump  = [&](size_t bytes)->char* { char* r = wp; wp += (bytes + 255) & ~((size_t)255); return r; };
  auto bumpU = [&](size_t e)->unsigned short* { return (unsigned short*)bump(e*2); };
  auto bumpF = [&](size_t e)->float* { return (float*)bump(e*4); };

  unsigned short* w_enc1 = bumpU(512*256);
  unsigned short* w_enc2 = bumpU(512*512);
  unsigned short* w_kf2  = bumpU(512*512);
  unsigned short* w_rel2 = bumpU(64*512);
  unsigned short* w_dec1 = bumpU(512*512);
  unsigned short* w_dec2 = bumpU(221*512);
  unsigned short* ET     = bumpU(64*1024);
  unsigned short *wq[NLn], *wk[NLn], *wv[NLn], *wo[NLn], *wf1[NLn], *wf2[NLn];
  for (int l = 0; l < NLn; l++) {
    wq[l]  = bumpU(512*512);  wk[l]  = bumpU(512*512);
    wv[l]  = bumpU(512*512);  wo[l]  = bumpU(512*512);
    wf1[l] = bumpU((size_t)2048*512); wf2[l] = bumpU((size_t)512*2048);
  }
  float* xcat  = bumpF((size_t)ROWS*256);
  float* kfh1  = bumpF((size_t)ROWS*512);
  float* pekf  = bumpF((size_t)ROWS*512);
  float* h     = bumpF((size_t)ROWS*512);
  float* hn    = bumpF((size_t)ROWS*512);
  float* regA  = bumpF((size_t)ROWS*2048);        // q|k|v|ao, later ffh
  float* q  = regA;
  float* k  = regA + (size_t)ROWS*512;
  float* v  = regA + (size_t)ROWS*1024;
  float* ao = regA + (size_t)ROWS*1536;
  float* ffh = regA;
  float* relh1 = bumpF((size_t)1024*512);
  float* erel  = bumpF((size_t)1024*64);
  float* dech1 = kfh1;   // dead after encoder
  float* ydec  = pekf;   // dead after encoder

  auto cvt = [&](int widx, unsigned short* dst, int N, int K, int Kp) {
    int n = N*Kp;
    cvt_w_kernel<<<(n+255)/256, 256, 0, stream>>>(F(widx), dst, N, K, Kp);
  };
  cvt(10, w_enc1, 512, 218, 256);
  cvt(12, w_enc2, 512, 512, 512);
  cvt(17, w_kf2,  512, 512, 512);
  cvt(REL_L2W, w_rel2, 64, 512, 512);
  cvt(5,  w_dec1, 512, 512, 512);
  cvt(7,  w_dec2, 221, 512, 512);
  for (int l = 0; l < NLn; l++) {
    int b0 = L0 + l*LS;
    cvt(b0+5,  wq[l],  512, 512, 512);
    cvt(b0+1,  wk[l],  512, 512, 512);
    cvt(b0+7,  wv[l],  512, 512, 512);
    cvt(b0+3,  wo[l],  512, 512, 512);
    cvt(b0+10, wf1[l], 2048, 512, 512);
    cvt(b0+12, wf2[l], 512, 2048, 2048);
  }

  auto gemm = [&](const float* X, int ldx, const unsigned short* W, const float* bias,
                  float* Y, int ldy, const float* add, int ldadd,
                  const float* alpha, int act, int M, int N, int Kp) {
    dim3 grid((N+63)/64, M/128);
    gemm_kernel<<<grid, 256, 0, stream>>>(X, ldx, W, bias, Y, ldy, add, ldadd, alpha, act, M, N, Kp);
  };

  // encoder + keyframe embedding
  concat_kernel<<<(ROWS*256+255)/256, 256, 0, stream>>>(x, maskin, xcat);
  kf1_kernel<<<(ROWS*DMn+255)/256, 256, 0, stream>>>(pkf, F(15), F(14), F(13), kfh1);
  gemm(kfh1, 512, w_kf2, F(16), pekf, 512, nullptr, 0, nullptr, 0, ROWS, 512, 512);
  gemm(xcat, 256, w_enc1, F(9),  hn, 512, nullptr, 0, F(8), 1, ROWS, 512, 256);
  gemm(hn,   512, w_enc2, F(11), h,  512, pekf, 512, F(8), 1, ROWS, 512, 512);

  // relative-position table E_rel -> ET (bf16, transposed, padded)
  rel1_kernel<<<(1024*DMn+255)/256, 256, 0, stream>>>(F(REL_L1W), F(REL_L1B), F(REL_A), relh1);
  gemm(relh1, 512, w_rel2, F(REL_L2B), erel, 64, nullptr, 0, nullptr, 0, 1024, 64, 512);
  build_et_kernel<<<(64*1024+255)/256, 256, 0, stream>>>(erel, ET);

  // transformer layers
  for (int l = 0; l < NLn; l++) {
    int b0 = L0 + l*LS;
    ln_kernel<<<ROWS/8, 256, 0, stream>>>(h, F(LN_G), F(LN_B), hn, ROWS);
    gemm(hn, 512, wq[l], F(b0+4), q, 512, nullptr, 0, nullptr, 0, ROWS, 512, 512);
    gemm(hn, 512, wk[l], F(b0+0), k, 512, nullptr, 0, nullptr, 0, ROWS, 512, 512);
    gemm(hn, 512, wv[l], F(b0+6), v, 512, nullptr, 0, nullptr, 0, ROWS, 512, 512);
    attn_kernel<<<dim3(BB*NHn, TTn/128), 256, 0, stream>>>(q, k, v, ET, maskin, ao);
    gemm(ao, 512, wo[l], F(b0+2), h, 512, h, 512, nullptr, 0, ROWS, 512, 512);
    ln_kernel<<<ROWS/8, 256, 0, stream>>>(h, F(LN_G), F(LN_B), hn, ROWS);
    gemm(hn, 512,  wf1[l], F(b0+9),  ffh, 2048, nullptr, 0, F(b0+8), 1, ROWS, 2048, 512);
    gemm(ffh, 2048, wf2[l], F(b0+11), h,   512,  h, 512, nullptr, 0, ROWS, 512, 2048);
  }

  // decoder + sigmoid split
  gemm(h,     512, w_dec1, F(4), dech1, 512, nullptr, 0, F(3), 1, ROWS, 512, 512);
  gemm(dech1, 512, w_dec2, F(6), ydec,  221, nullptr, 0, nullptr, 0, ROWS, 221, 512);
  final_kernel<<<(ROWS*221+255)/256, 256, 0, stream>>>(ydec, (float*)d_out);
}
